// GAT_75393855914352
// MI455X (gfx1250) — matrix operand order repaired and checked by reading, not compiled, not run on hardware
//
#include <hip/hip_runtime.h>
#include <hip/hip_bf16.h>
#include <math.h>

// ---------------------------------------------------------------------------
// Types for CDNA5 WMMA
// ---------------------------------------------------------------------------
typedef __attribute__((ext_vector_type(16))) __bf16 v16bf;
typedef __attribute__((ext_vector_type(8)))  __bf16 v8bf;
typedef __attribute__((ext_vector_type(8)))  float  v8f;
typedef __attribute__((ext_vector_type(4)))  float  v4f;

static __device__ __forceinline__ __bf16 f32_to_bf16(float f) {
    // round-to-nearest-even bf16 conversion via bit manipulation
    unsigned u = __float_as_uint(f);
    unsigned r = (u + 0x7FFFu + ((u >> 16) & 1u)) >> 16;
    unsigned short h = (unsigned short)r;
    return __builtin_bit_cast(__bf16, h);
}

// ---------------------------------------------------------------------------
// bf16 WMMA GEMM: C[M,Nc] = A[M,K] @ B[K,Nc]  (A,B f32 in memory; bf16
// fragments, f32 accumulate).
//  - B panel (K x 64) staged once per block into LDS as bf16, transposed
//    [col][k] with +8 padding -> each lane's fragment is 2x ds_load_b128.
//  - A loaded as float4 (global_load_b128), converted in registers.
//  - One wave -> 16x64 strip: 4 accumulators share one A fragment ->
//    4 back-to-back v_wmma_f32_16x16x32_bf16 per K-step.
// Requires M % 16 == 0 (N = 100000 = 6250*16), Nc % 64 == 0 (64, 320),
// K % 32 == 0 (512, 64).
// ---------------------------------------------------------------------------
#define GEMM_WAVES_PER_BLOCK 8
#define GEMM_NTILES 4
#define BPAD 8   // bf16 elements of padding per LDS column (16B -> 4 banks)

__global__ void gat_wmma_gemm(const float* __restrict__ A,
                              const float* __restrict__ B,
                              float* __restrict__ C,
                              int M, int K, int Nc) {
    extern __shared__ __bf16 Blds[];          // 64 * (K + BPAD) bf16

    const int lane = threadIdx.x & 31;
    const int wave = threadIdx.x >> 5;
    const int tm = blockIdx.x * GEMM_WAVES_PER_BLOCK + wave;  // 16-row tile
    const int tn = blockIdx.y;                                // 64-col group
    const int Kp = K + BPAD;

    // ---- stage B panel into LDS (all waves participate; coalesced reads) --
    for (int idx = threadIdx.x; idx < K * 64; idx += blockDim.x) {
        const int col = idx & 63;             // fastest -> coalesced global
        const int k   = idx >> 6;
        Blds[col * Kp + k] = f32_to_bf16(B[(size_t)k * Nc + tn * 64 + col]);
    }
    __syncthreads();

    if (tm * 16 >= M) return;                 // wave-uniform (EXEC stays full)

    const int half = lane >> 4;               // 0 or 1 (lane group)
    const int row  = tm * 16 + (lane & 15);
    const int khb  = half * 16;               // B: 16 K-rows per lane-half
    const int kha  = half * 16;               // A: 16 K-cols per lane-half
    const __bf16* bcol = Blds + (lane & 15) * Kp;

    v8f acc[GEMM_NTILES] = {};
    const float* Ar = A + (size_t)row * K;

    for (int k0 = 0; k0 < K; k0 += 32) {
        if (k0 + 32 < K)                      // prefetch next A K-block
            __builtin_prefetch(&Ar[k0 + 32 + kha], 0, 1);

        // A fragment: 16 bf16 from 4x float4 loads
        v4f a0 = *(const v4f*)(Ar + k0 + kha);
        v4f a1 = *(const v4f*)(Ar + k0 + kha + 4);
        v4f a2 = *(const v4f*)(Ar + k0 + kha + 8);
        v4f a3 = *(const v4f*)(Ar + k0 + kha + 12);
        v16bf a;
#pragma unroll
        for (int j = 0; j < 4; ++j) {
            a[j]      = f32_to_bf16(a0[j]);
            a[j + 4]  = f32_to_bf16(a1[j]);
            a[j + 8]  = f32_to_bf16(a2[j]);
            a[j + 12] = f32_to_bf16(a3[j]);
        }

#pragma unroll
        for (int t = 0; t < GEMM_NTILES; ++t) {
            // B fragment: 16 contiguous bf16 in LDS -> 2x ds_load_b128
            const __bf16* bp = bcol + t * 16 * Kp + k0 + khb;
            v8bf blo = *(const v8bf*)bp;
            v8bf bhi = *(const v8bf*)(bp + 8);
            v16bf b;
#pragma unroll
            for (int j = 0; j < 8; ++j) { b[j] = blo[j]; b[j + 8] = bhi[j]; }
            acc[t] = __builtin_amdgcn_wmma_f32_16x16x32_bf16(
                false, a, false, b, (short)0, acc[t], false, false);
        }
    }

    // C layout: VGPR r -> M = r + half*8 ; N = lane&15 (+ tile offset)
    const int mrow = tm * 16 + half * 8;
    float* Cr = C + (size_t)mrow * Nc + tn * 64 + (lane & 15);
#pragma unroll
    for (int t = 0; t < GEMM_NTILES; ++t)
#pragma unroll
        for (int r = 0; r < 8; ++r)
            Cr[(size_t)r * Nc + t * 16] = acc[t][r];
}

// ---------------------------------------------------------------------------
// Per-node attention logits: as[n,h] = sum_c h[n,h,c]*a_src[h,c]; same for ad
// ---------------------------------------------------------------------------
__global__ void gat_node_alpha(const float* __restrict__ h,
                               const float* __restrict__ a_src,
                               const float* __restrict__ a_dst,
                               float* __restrict__ as, float* __restrict__ ad,
                               int N, int H, int C) {
    int i = blockIdx.x * blockDim.x + threadIdx.x;
    if (i >= N * H) return;
    const int n = i / H, hh = i % H;
    const float* hp = h + (long)n * H * C + (long)hh * C;
    float s = 0.f, d = 0.f;
    for (int c = 0; c < C; ++c) {
        float v = hp[c];
        s += v * a_src[hh * C + c];
        d += v * a_dst[hh * C + c];
    }
    as[i] = s;
    ad[i] = d;
}

// ---------------------------------------------------------------------------
// Edge helpers
// ---------------------------------------------------------------------------
static __device__ __forceinline__ void edge_sd(const long long* __restrict__ ei,
                                               long long E, long long e,
                                               int& s, int& d) {
    if (e < E) { s = (int)ei[e]; d = (int)ei[E + e]; }
    else       { s = d = (int)(e - E); }          // appended self-loop
}

static __device__ __forceinline__ float leaky(float v) {
    return v > 0.f ? v : 0.2f * v;
}

static __device__ __forceinline__ void atomicMaxF(float* addr, float v) {
    // int-ordered float max: correct for mixed signs with -inf init
    if (v >= 0.f) atomicMax((int*)addr, __float_as_int(v));
    else          atomicMin((unsigned int*)addr, __float_as_uint(v));
}

__global__ void gat_edge_max(const long long* __restrict__ ei, long long E,
                             long long Etot, const float* __restrict__ as,
                             const float* __restrict__ ad,
                             float* __restrict__ m, int H) {
    long long e = blockIdx.x * (long long)blockDim.x + threadIdx.x;
    if (e >= Etot) return;
    int s, d; edge_sd(ei, E, e, s, d);
#pragma unroll
    for (int hh = 0; hh < 8; ++hh) {
        float v = leaky(as[s * H + hh] + ad[d * H + hh]);
        atomicMaxF(&m[d * H + hh], v);
    }
}

__global__ void gat_edge_sumexp(const long long* __restrict__ ei, long long E,
                                long long Etot, const float* __restrict__ as,
                                const float* __restrict__ ad,
                                const float* __restrict__ m,
                                float* __restrict__ sums, int H) {
    long long e = blockIdx.x * (long long)blockDim.x + threadIdx.x;
    if (e >= Etot) return;
    int s, d; edge_sd(ei, E, e, s, d);
#pragma unroll
    for (int hh = 0; hh < 8; ++hh) {
        float v = leaky(as[s * H + hh] + ad[d * H + hh]);
        atomicAdd(&sums[d * H + hh], expf(v - m[d * H + hh]));
    }
}

// Scatter for concat layers (out[dst,h,c] += h[src,h,c]*alpha)
__global__ void gat_edge_scatter(const long long* __restrict__ ei, long long E,
                                 long long Etot, const float* __restrict__ h,
                                 const float* __restrict__ as,
                                 const float* __restrict__ ad,
                                 const float* __restrict__ m,
                                 const float* __restrict__ sums,
                                 float* __restrict__ out, int H, int C) {
    long long e = blockIdx.x * (long long)blockDim.x + threadIdx.x;
    if (e >= Etot) return;
    int s, d; edge_sd(ei, E, e, s, d);
#pragma unroll
    for (int hh = 0; hh < 8; ++hh) {
        float v = leaky(as[s * H + hh] + ad[d * H + hh]);
        float alpha = expf(v - m[d * H + hh]) / (sums[d * H + hh] + 1e-16f);
        const float* hp = h + (long)s * H * C + (long)hh * C;
        float* op = out + (long)d * H * C + (long)hh * C;
        for (int c = 0; c < C; ++c)
            atomicAdd(&op[c], hp[c] * alpha);
    }
}

// Scatter for final layer with fused head-mean: out[dst,c] += mean_h(...)
__global__ void gat_edge_scatter_mean(const long long* __restrict__ ei,
                                      long long E, long long Etot,
                                      const float* __restrict__ h,
                                      const float* __restrict__ as,
                                      const float* __restrict__ ad,
                                      const float* __restrict__ m,
                                      const float* __restrict__ sums,
                                      float* __restrict__ out, int H, int C) {
    long long e = blockIdx.x * (long long)blockDim.x + threadIdx.x;
    if (e >= Etot) return;
    int s, d; edge_sd(ei, E, e, s, d);
    float alpha[8];
#pragma unroll
    for (int hh = 0; hh < 8; ++hh) {
        float v = leaky(as[s * H + hh] + ad[d * H + hh]);
        alpha[hh] = expf(v - m[d * H + hh]) / (sums[d * H + hh] + 1e-16f);
    }
    const float* hp = h + (long)s * H * C;
    float* op = out + (long)d * C;
    for (int c = 0; c < C; ++c) {
        float accv = 0.f;
#pragma unroll
        for (int hh = 0; hh < 8; ++hh)
            accv += hp[hh * C + c] * alpha[hh];
        atomicAdd(&op[c], accv * 0.125f);
    }
}

// ---------------------------------------------------------------------------
// Epilogues + init
// ---------------------------------------------------------------------------
__global__ void gat_bias_elu(float* __restrict__ out,
                             const float* __restrict__ b,
                             long long n, int HC) {
    long long i = blockIdx.x * (long long)blockDim.x + threadIdx.x;
    if (i >= n) return;
    float v = out[i] + b[(int)(i % HC)];
    out[i] = v > 0.f ? v : (expf(v) - 1.f);
}

__global__ void gat_final_lsm(const float* __restrict__ acc,
                              const float* __restrict__ b2,
                              float* __restrict__ out, int N, int C) {
    int n = blockIdx.x * blockDim.x + threadIdx.x;
    if (n >= N) return;
    float vals[40];
    float mx = -INFINITY;
    for (int c = 0; c < C; ++c) {
        float v = acc[(long)n * C + c] + b2[c];
        vals[c] = v;
        mx = fmaxf(mx, v);
    }
    float s = 0.f;
    for (int c = 0; c < C; ++c) s += expf(vals[c] - mx);
    float lse = logf(s) + mx;
    for (int c = 0; c < C; ++c) out[(long)n * C + c] = vals[c] - lse;
}

__global__ void gat_fill(float* __restrict__ p, float v, long long n) {
    long long i = blockIdx.x * (long long)blockDim.x + threadIdx.x;
    if (i < n) p[i] = v;
}

// ---------------------------------------------------------------------------
// Orchestration
// ---------------------------------------------------------------------------
extern "C" void kernel_launch(void* const* d_in, const int* in_sizes, int n_in,
                              void* d_out, int out_size, void* d_ws, size_t ws_size,
                              hipStream_t stream) {
    (void)n_in; (void)out_size; (void)ws_size;

    const float*      x   = (const float*)d_in[0];
    const long long*  ei  = (const long long*)d_in[1];
    const float* W0 = (const float*)d_in[2];
    const float* asrc0 = (const float*)d_in[3];
    const float* adst0 = (const float*)d_in[4];
    const float* b0 = (const float*)d_in[5];
    const float* W1 = (const float*)d_in[6];
    const float* asrc1 = (const float*)d_in[7];
    const float* adst1 = (const float*)d_in[8];
    const float* b1 = (const float*)d_in[9];
    const float* W2 = (const float*)d_in[10];
    const float* asrc2 = (const float*)d_in[11];
    const float* adst2 = (const float*)d_in[12];
    const float* b2 = (const float*)d_in[13];

    const int F_IN = 512, H = 8, C_HID = 8, N_CLS = 40;
    const int HID = H * C_HID;                 // 64
    const int N = in_sizes[0] / F_IN;          // 100000
    const long long E = in_sizes[1] / 2;       // 1.6M
    const long long Etot = E + N;              // + self-loops

    // workspace partition (floats): bufH [N*320] | bufO [N*64] | as,ad,m,s [N*8]x4
    float* bufH = (float*)d_ws;
    float* bufO = bufH + (size_t)N * 320;
    float* asb  = bufO + (size_t)N * 64;
    float* adb  = asb + (size_t)N * H;
    float* mb   = adb + (size_t)N * H;
    float* sb   = mb  + (size_t)N * H;

    const int TB = 256;
    const int mtiles = (N + 15) / 16;
    const int gmx = (mtiles + GEMM_WAVES_PER_BLOCK - 1) / GEMM_WAVES_PER_BLOCK;
    dim3 blk(TB);
    int nhBlocks  = (N * H + TB - 1) / TB;
    int egBlocks  = (int)((Etot + TB - 1) / TB);
    int nodeBlocks = (N + TB - 1) / TB;

    auto gemm_shmem = [&](int K) { return (size_t)64 * (K + BPAD) * sizeof(__hip_bfloat16); };

    auto run_edge_softmax = [&](const float* hfeat, int C, float* outbuf,
                                long long outElems, bool meanHeads) {
        gat_fill<<<(int)((N * (long long)H + TB - 1) / TB), blk, 0, stream>>>(mb, -INFINITY, (long long)N * H);
        gat_fill<<<(int)((N * (long long)H + TB - 1) / TB), blk, 0, stream>>>(sb, 0.f, (long long)N * H);
        gat_fill<<<(int)((outElems + TB - 1) / TB), blk, 0, stream>>>(outbuf, 0.f, outElems);
        gat_edge_max<<<egBlocks, blk, 0, stream>>>(ei, E, Etot, asb, adb, mb, H);
        gat_edge_sumexp<<<egBlocks, blk, 0, stream>>>(ei, E, Etot, asb, adb, mb, sb, H);
        if (meanHeads)
            gat_edge_scatter_mean<<<egBlocks, blk, 0, stream>>>(ei, E, Etot, hfeat, asb, adb, mb, sb, outbuf, H, C);
        else
            gat_edge_scatter<<<egBlocks, blk, 0, stream>>>(ei, E, Etot, hfeat, asb, adb, mb, sb, outbuf, H, C);
    };

    // ---------------- Layer 0: x[N,512] @ W0 -> h0[N,64] ----------------
    gat_wmma_gemm<<<dim3(gmx, HID / 64), blk, gemm_shmem(F_IN), stream>>>(x, W0, bufH, N, F_IN, HID);
    gat_node_alpha<<<nhBlocks, blk, 0, stream>>>(bufH, asrc0, adst0, asb, adb, N, H, C_HID);
    run_edge_softmax(bufH, C_HID, bufO, (long long)N * HID, false);
    gat_bias_elu<<<(int)(((long long)N * HID + TB - 1) / TB), blk, 0, stream>>>(bufO, b0, (long long)N * HID, HID);

    // ---------------- Layer 1: h[N,64] @ W1 -> h1[N,64] ----------------
    gat_wmma_gemm<<<dim3(gmx, HID / 64), blk, gemm_shmem(HID), stream>>>(bufO, W1, bufH, N, HID, HID);
    gat_node_alpha<<<nhBlocks, blk, 0, stream>>>(bufH, asrc1, adst1, asb, adb, N, H, C_HID);
    run_edge_softmax(bufH, C_HID, bufO, (long long)N * HID, false);
    gat_bias_elu<<<(int)(((long long)N * HID + TB - 1) / TB), blk, 0, stream>>>(bufO, b1, (long long)N * HID, HID);

    // ---------------- Layer 2: h[N,64] @ W2 -> h2[N,320], mean heads ----
    gat_wmma_gemm<<<dim3(gmx, (H * N_CLS) / 64), blk, gemm_shmem(HID), stream>>>(bufO, W2, bufH, N, HID, H * N_CLS);
    gat_node_alpha<<<nhBlocks, blk, 0, stream>>>(bufH, asrc2, adst2, asb, adb, N, H, N_CLS);
    run_edge_softmax(bufH, N_CLS, bufO, (long long)N * N_CLS, true);
    gat_final_lsm<<<nodeBlocks, blk, 0, stream>>>(bufO, b2, (float*)d_out, N, N_CLS);
}
